// ExodusModel_90452011254006
// MI455X (gfx1250) — matrix operand order repaired and checked by reading, not compiled, not run on hardware
//
#include <hip/hip_runtime.h>

typedef _Float16 h16;
typedef __attribute__((ext_vector_type(4)))  _Float16 v4h;
typedef __attribute__((ext_vector_type(16))) _Float16 v16h;
typedef __attribute__((ext_vector_type(8)))  float    v8f;

#define T 100   // timesteps
#define NN 8    // batch

// ---------------------------------------------------------------------------
// Kernel 1: avg_pool4 of raw input, f32 -> f16.
// in : [8][2][256][256][100] f32   out: [8][2][64][64][100] f16
// One block per (n,c,oy,ox); lane = t (coalesced along innermost t).
// Only kernel touching the 419 MB input; NT loads keep L2 free for the
// L2-resident intermediates used by every later stage.
// ---------------------------------------------------------------------------
__global__ void pool0_kernel(const float* __restrict__ in, h16* __restrict__ out) {
  int b = blockIdx.x;
  int t = threadIdx.x;
  if (t >= T) return;
  int ox = b & 63; b >>= 6;
  int oy = b & 63; b >>= 6;
  int c  = b & 1;  b >>= 1;
  int n  = b;
  const float* src = in + ((((size_t)n*2 + c)*256 + oy*4)*256 + ox*4)*T + t;
  float s = 0.f;
  #pragma unroll
  for (int dy = 0; dy < 4; ++dy)
    #pragma unroll
    for (int dx = 0; dx < 4; ++dx)
      s += __builtin_nontemporal_load(&src[(size_t)(dy*256 + dx)*T]);
  out[((((size_t)n*2 + c)*64 + oy)*64 + ox)*T + t] = (h16)(s * 0.0625f);
}

// ---------------------------------------------------------------------------
// Zero page: 128 zero halves; padded-K / out-of-image gather elements are
// redirected here so the conv inner loop needs no predication.
// ---------------------------------------------------------------------------
__global__ void zero_page_kernel(h16* __restrict__ zp) {
  zp[threadIdx.x] = (h16)0.f;
}

// ---------------------------------------------------------------------------
// Weight prepack into WMMA A fragments (this code's contiguous k order, same as B):
// A is 16x32 (MxK); lane l holds row M=l&15; element e of v16h holds
// K = 16*(l>>4) + e.  Chunk c covers K global [32c,32c+32).
// pk element index = (c*32 + lane)*16 + e  (wave loads contiguous 32B/lane).
// ---------------------------------------------------------------------------
__global__ void pack_w_kernel(const float* __restrict__ w, h16* __restrict__ pk,
                              int cin, int cout, int kch) {
  int i = blockIdx.x * blockDim.x + threadIdx.x;
  if (i >= kch * 512) return;
  int e = i & 15;
  int l = (i >> 4) & 31;
  int c = i >> 9;
  int m  = l & 15;
  int kl = 16*(l >> 4) + e;
  int k  = c*32 + kl;
  float v = 0.f;
  if (m < cout && k < cin*49) {
    int ic = k / 49, r = k % 49;
    v = w[((m*cin + ic)*7 + r/7)*7 + (r % 7)];
  }
  pk[(size_t)(c*32 + l)*16 + e] = (h16)v;
}

// ---------------------------------------------------------------------------
// 7x7 SAME conv as WMMA GEMM.  One wave owns one (n,pixel), loops 7 t-tiles.
//   D[16x16] = sum_c A_c(weights 16xK) x B_c(im2col patches Kx16)
//   M = out-channel (COUT valid, padded 16)
//   N = 16 consecutive timesteps (t innermost -> contiguous 32B B rows)
//   K = CIN*49 padded to KCH*32
// Phase 1: wave resolves all K gather offsets once into LDS (invalid -> zero
// page), removing all decode/bounds work from the t loop.
// Phase 2: per tile/chunk: 4x ds_load_b128 offsets + 16 unconditional
// global_load_d16 + v_wmma_f32_16x16x32_f16, f32 accumulate.
// B layout (16-bit B KxN): lane l holds col N=l&15, element e -> K=16*(l>>4)+e.
// Columns tcol>=100 read a few elements past a row (still-mapped finite f16
// ws data) and are never stored.
// ---------------------------------------------------------------------------
template<int CIN, int COUT, int S, int KCH>
__global__ void conv_wmma_kernel(const h16* __restrict__ img,   // buffer base
                                 const h16* __restrict__ pw,
                                 float* __restrict__ z,
                                 int zpIdx) {                    // zero page idx rel. img
  __shared__ int offT[8][KCH*32];
  int tid  = threadIdx.x;
  int lane = tid & 31;
  int wv   = tid >> 5;
  int wave = blockIdx.x * (blockDim.x >> 5) + wv;
  int p  = wave % (S*S);
  int n  = wave / (S*S);
  int py = p / S, px = p % S;

  // Phase 1: resolve gather offsets (element index of img[n][ic][yy][xx][0]).
  for (int k = lane; k < KCH*32; k += 32) {
    int off = zpIdx;
    if (k < CIN*49) {
      int ic = k / 49, r = k % 49;
      int yy = py + r/7 - 3;
      int xx = px + (r % 7) - 3;
      if ((unsigned)yy < (unsigned)S && (unsigned)xx < (unsigned)S)
        off = (((n*CIN + ic)*S + yy)*S + xx)*T;
    }
    offT[wv][k] = off;
  }
  __syncthreads();

  int hh = lane >> 4;    // lane half selects K sub-range in fragments
  int tl = lane & 15;    // N column (timestep within tile)

  // Prepacked weights: contiguous 32 B per lane per chunk.
  v16h a[KCH];
  #pragma unroll
  for (int c = 0; c < KCH; ++c)
    a[c] = *(const v16h*)(pw + (size_t)(c*32 + lane)*16);

  const int* myOff = &offT[wv][hh*16];

  for (int tt = 0; tt < 7; ++tt) {        // 7 tiles cover T=100 (last partial)
    int tcol = tt*16 + tl;
    v8f acc = {};
    #pragma unroll
    for (int c = 0; c < KCH; ++c) {
      int off[16];
      #pragma unroll
      for (int j = 0; j < 4; ++j)         // 16B-aligned LDS vector reads
        *(int4*)&off[j*4] = *(const int4*)&myOff[c*32 + j*4];
      v16h b;
      #pragma unroll
      for (int e = 0; e < 16; ++e)
        b[e] = img[(size_t)(off[e] + tcol)];
      acc = __builtin_amdgcn_wmma_f32_16x16x32_f16(
          false, a[c], false, b, (short)0, acc, false, false);
    }
    // C/D layout: VGPR r, lanes 0-15 -> row M=r, col N=lane. Valid oc < COUT.
    if (hh == 0 && tcol < T) {
      #pragma unroll
      for (int r = 0; r < COUT; ++r)
        z[((((size_t)n*COUT + r)*S + py)*S + px)*T + tcol] = acc[r];
    }
  }
}

// ---------------------------------------------------------------------------
// IAF scan: v += z_t; spike = (v >= 1); v -= spike.  One thread per chain,
// 100 contiguous f32 in / 100 contiguous f16 out.  The recurrence is a true
// serial dep; memory ops are 4x vectorized (float4 in, 4xf16 out).
// ---------------------------------------------------------------------------
__global__ void iaf_kernel(const float* __restrict__ z, h16* __restrict__ s,
                           int chains) {
  int i = blockIdx.x * blockDim.x + threadIdx.x;
  if (i >= chains) return;
  const float* zp = z + (size_t)i * T;
  h16*         sp = s + (size_t)i * T;
  float v = 0.f;
  for (int tb = 0; tb < T; tb += 4) {
    float4 zz = *(const float4*)&zp[tb];
    v4h out;
    float sk;
    v += zz.x; sk = (v >= 1.f) ? 1.f : 0.f; v -= sk; out[0] = (h16)sk;
    v += zz.y; sk = (v >= 1.f) ? 1.f : 0.f; v -= sk; out[1] = (h16)sk;
    v += zz.z; sk = (v >= 1.f) ? 1.f : 0.f; v -= sk; out[2] = (h16)sk;
    v += zz.w; sk = (v >= 1.f) ? 1.f : 0.f; v -= sk; out[3] = (h16)sk;
    *(v4h*)&sp[tb] = out;
  }
}

// ---------------------------------------------------------------------------
// avg_pool4 on f16 spike maps (spatial SxS -> S/4 x S/4), t innermost.
// Spike means k/16 are exactly representable in f16.
// ---------------------------------------------------------------------------
template<int S, int C>
__global__ void pool4h_kernel(const h16* __restrict__ in, h16* __restrict__ out) {
  const int OS = S / 4;
  int b = blockIdx.x;
  int t = threadIdx.x;
  if (t >= T) return;
  int ox = b % OS; b /= OS;
  int oy = b % OS; b /= OS;
  int c  = b % C;
  int n  = b / C;
  const h16* src = in + ((((size_t)(n*C + c))*S + oy*4)*S + ox*4)*T + t;
  float s = 0.f;
  #pragma unroll
  for (int dy = 0; dy < 4; ++dy)
    #pragma unroll
    for (int dx = 0; dx < 4; ++dx)
      s += (float)src[(size_t)(dy*S + dx)*T];
  out[((((size_t)(n*C + c))*OS + oy)*OS + ox)*T + t] = (h16)(s * 0.0625f);
}

// ---------------------------------------------------------------------------
// Head: avg_pool4 of s1 (8,16,16) -> 128 features, then 128->2 linear.
// Tiny (800 rows); one thread per (n,t). out[n][cls][t] f32.
// ---------------------------------------------------------------------------
__global__ void head_kernel(const h16* __restrict__ s1,
                            const float* __restrict__ w,
                            float* __restrict__ out) {
  int n = blockIdx.x;
  int t = threadIdx.x;
  if (t >= T) return;
  float a0 = 0.f, a1 = 0.f;
  for (int oc = 0; oc < 8; ++oc)
    for (int yy = 0; yy < 4; ++yy)
      for (int xx = 0; xx < 4; ++xx) {
        float f = 0.f;
        #pragma unroll
        for (int dy = 0; dy < 4; ++dy)
          #pragma unroll
          for (int dx = 0; dx < 4; ++dx)
            f += (float)s1[((((size_t)(n*8 + oc))*16 + yy*4 + dy)*16 + xx*4 + dx)*T + t];
        f *= 0.0625f;
        int kf = oc*16 + yy*4 + xx;       // reshape order of (c=8,h=4,w=4)
        a0 += f * w[kf];
        a1 += f * w[128 + kf];
      }
  out[(size_t)n*200 + t]       = a0;     // (n, 2, t) layout
  out[(size_t)n*200 + 100 + t] = a1;
}

// ---------------------------------------------------------------------------
// Launch
// ---------------------------------------------------------------------------
static inline size_t alignup(size_t x) { return (x + 255) & ~(size_t)255; }

extern "C" void kernel_launch(void* const* d_in, const int* in_sizes, int n_in,
                              void* d_out, int out_size, void* d_ws, size_t ws_size,
                              hipStream_t stream) {
  (void)in_sizes; (void)n_in; (void)out_size; (void)ws_size;
  const float* inp = (const float*)d_in[0];   // (8,2,256,256,100)
  const float* w0  = (const float*)d_in[1];   // (4,2,7,7)
  const float* w1  = (const float*)d_in[2];   // (8,4,7,7)
  const float* wl  = (const float*)d_in[3];   // (2,128)
  float* out = (float*)d_out;                 // (8,2,100)

  // Workspace carve-up (~106 MB; all intermediates fit the 192 MB L2).
  char* ws = (char*)d_ws;
  size_t o = 0;
  h16*   p0  = (h16*)(ws + o); o = alignup(o + (size_t)NN*2*64*64*T*2);   // pooled0
  h16*   pw0 = (h16*)(ws + o); o = alignup(o + (size_t)4*512*2);          // packed W0
  float* z0  = (float*)(ws + o); o = alignup(o + (size_t)NN*4*64*64*T*4); // conv0 currents
  h16*   s0  = (h16*)(ws + o); o = alignup(o + (size_t)NN*4*64*64*T*2);   // spikes0
  h16*   p1  = (h16*)(ws + o); o = alignup(o + (size_t)NN*4*16*16*T*2);   // pooled1
  h16*   pw1 = (h16*)(ws + o); o = alignup(o + (size_t)7*512*2);          // packed W1
  float* z1  = (float*)(ws + o); o = alignup(o + (size_t)NN*8*16*16*T*4); // conv1 currents
  h16*   s1  = (h16*)(ws + o); o = alignup(o + (size_t)NN*8*16*16*T*2);   // spikes1
  h16*   zp  = (h16*)(ws + o); o = alignup(o + (size_t)128*2);            // zero page

  int zpIdx0 = (int)(zp - p0);   // zero-page element index relative to p0
  int zpIdx1 = (int)(zp - p1);   // ... and relative to p1 (both positive)

  // 1) HBM pass: pool the raw input, convert to f16 (NT loads).
  pool0_kernel<<<NN*2*64*64, 128, 0, stream>>>(inp, p0);

  // 2) Zero page + prepack conv weights into WMMA A-fragment layout.
  zero_page_kernel<<<1, 128, 0, stream>>>(zp);
  pack_w_kernel<<<(4*512 + 255)/256, 256, 0, stream>>>(w0, pw0, 2, 4, 4);
  pack_w_kernel<<<(7*512 + 255)/256, 256, 0, stream>>>(w1, pw1, 4, 8, 7);

  // 3) conv0 as WMMA: 8 waves/block, one wave per (n,pixel). 32768 waves.
  conv_wmma_kernel<2, 4, 64, 4><<<NN*64*64/8, 256, 0, stream>>>(p0, pw0, z0, zpIdx0);

  // 4) IAF scan layer 0: 131072 chains.
  iaf_kernel<<<(NN*4*64*64 + 255)/256, 256, 0, stream>>>(z0, s0, NN*4*64*64);

  // 5) pool spikes0 -> pooled1.
  pool4h_kernel<64, 4><<<NN*4*16*16, 128, 0, stream>>>(s0, p1);

  // 6) conv1 as WMMA: 2048 waves.
  conv_wmma_kernel<4, 8, 16, 7><<<NN*16*16/8, 256, 0, stream>>>(p1, pw1, z1, zpIdx1);

  // 7) IAF scan layer 1: 16384 chains.
  iaf_kernel<<<(NN*8*16*16 + 255)/256, 256, 0, stream>>>(z1, s1, NN*8*16*16);

  // 8) pool + linear head -> (8,2,100).
  head_kernel<<<NN, 128, 0, stream>>>(s1, wl, out);
}
